// MyGNN_49555332661652
// MI455X (gfx1250) — matrix operand order repaired and checked by reading, not compiled, not run on hardware
//
#include <hip/hip_runtime.h>
#include <hip/hip_bf16.h>
#include <math.h>

// ---------------- types ----------------
typedef __attribute__((ext_vector_type(16))) __bf16 v16bf;
typedef __attribute__((ext_vector_type(8)))  __bf16 v8bf;
typedef __attribute__((ext_vector_type(8)))  float  v8f;

#define BN_EPS 1e-5f

__device__ inline unsigned short f2bf(float f) {
    unsigned int u = __float_as_uint(f);
    unsigned int r = u + 0x7FFFu + ((u >> 16) & 1u);
    return (unsigned short)(r >> 16);
}

__device__ inline float hash_uniform(unsigned int i) {
    unsigned int x = i * 0x9E3779B9u + 0x7F4A7C15u;
    x ^= x >> 16; x *= 0x85EBCA6Bu;
    x ^= x >> 13; x *= 0xC2B2AE35u;
    x ^= x >> 16;
    float u = (float)(x >> 8) * (1.0f / 16777216.0f);
    return fminf(fmaxf(u, 1e-6f), 1.0f - 1e-6f);
}

// single-instruction relu: v_med3_f32(x, 0, BIG)
__device__ inline float relu1(float x) {
    return __builtin_amdgcn_fmed3f(x, 0.0f, 3.0e38f);
}

// ---------------- utility ----------------
__global__ void zero_f32(float* __restrict__ p, int n) {
    int i = blockIdx.x * blockDim.x + threadIdx.x;
    if (i < n) p[i] = 0.0f;
}

// ---------------- GIN ----------------
__global__ void scatter_add_kernel(const float* __restrict__ x, const int* __restrict__ ei,
                                   float* __restrict__ agg, int E, int din) {
    int idx = blockIdx.x * blockDim.x + threadIdx.x;
    if (idx >= E * din) return;
    int e = idx / din, f = idx - e * din;
    int s = ei[e];
    int d = ei[E + e];
    atomicAdd(&agg[(size_t)d * din + f], x[(size_t)s * din + f]);
}

__global__ void gin_lin1_kernel(const float* __restrict__ x, const float* __restrict__ agg,
                                const float* __restrict__ epsp, const float* __restrict__ W1,
                                const float* __restrict__ b1, float* __restrict__ out,
                                int n, int din, int dout) {
    int idx = blockIdx.x * blockDim.x + threadIdx.x;
    if (idx >= n * dout) return;
    int i = idx / dout, o = idx - i * dout;
    float eps1 = 1.0f + epsp[0];
    const float* xi = x + (size_t)i * din;
    const float* ai = agg + (size_t)i * din;
    float acc = b1[o];
    for (int d = 0; d < din; ++d)
        acc += (eps1 * xi[d] + ai[d]) * W1[(size_t)d * dout + o];
    out[(size_t)i * dout + o] = fmaxf(acc, 0.0f);
}

__global__ void gin_lin2_kernel(const float* __restrict__ h, const float* __restrict__ W2,
                                const float* __restrict__ b2, float* __restrict__ out,
                                float* __restrict__ stats, int n, int dmid, int dout) {
    int idx = blockIdx.x * blockDim.x + threadIdx.x;
    if (idx >= n * dout) return;
    int i = idx / dout, o = idx - i * dout;
    const float* hi = h + (size_t)i * dmid;
    float acc = b2[o];
    for (int d = 0; d < dmid; ++d)
        acc += hi[d] * W2[(size_t)d * dout + o];
    out[(size_t)i * dout + o] = acc;
    atomicAdd(&stats[o], acc);
    atomicAdd(&stats[dout + o], acc * acc);
}

__global__ void gin_bn_kernel(const float* __restrict__ h, const float* __restrict__ stats,
                              const float* __restrict__ gamma, const float* __restrict__ beta,
                              float* __restrict__ out, int n, int dout, int do_relu) {
    int idx = blockIdx.x * blockDim.x + threadIdx.x;
    if (idx >= n * dout) return;
    int i = idx / dout, o = idx - i * dout;
    float inv_n = 1.0f / (float)n;
    float mean = stats[o] * inv_n;
    float var = stats[dout + o] * inv_n - mean * mean;
    float v = gamma[o] * (h[(size_t)i * dout + o] - mean) * rsqrtf(var + BN_EPS) + beta[o];
    out[(size_t)i * dout + o] = do_relu ? fmaxf(v, 0.0f) : v;
}

// ---------------- attention pooling ----------------
__global__ void colsum_kernel(const float* __restrict__ a, int n, float* __restrict__ colsum) {
    int idx = blockIdx.x * blockDim.x + threadIdx.x;
    if (idx >= n * 32) return;
    int i = idx >> 5, e = idx & 31;
    atomicAdd(&colsum[e], a[(size_t)i * 32 + e]);
}

__global__ void ctx_kernel(const float* __restrict__ colsum, const float* __restrict__ W,
                           float inv_n, float* __restrict__ ctx) {
    int e = threadIdx.x;
    if (e >= 32) return;
    float acc = 0.0f;
    for (int d = 0; d < 32; ++d) acc += colsum[d] * W[d * 32 + e];
    ctx[e] = tanhf(acc * inv_n);
}

__global__ void pool_kernel(const float* __restrict__ a, const float* __restrict__ ctx,
                            int n, float* __restrict__ p) {
    int i = blockIdx.x * blockDim.x + threadIdx.x;
    if (i >= n) return;
    const float* ai = a + (size_t)i * 32;
    float dot = 0.0f;
    for (int e = 0; e < 32; ++e) dot += ai[e] * ctx[e];
    float s = 1.0f / (1.0f + expf(-dot));
    for (int e = 0; e < 32; ++e) atomicAdd(&p[e], ai[e] * s);
}

// ---------------- T1 = a1 @ W_k per channel, bf16 output (padded rows zeroed) ----------------
__global__ void t1_kernel(const float* __restrict__ a, const float* __restrict__ W,
                          int n, int npad, unsigned short* __restrict__ out) {
    int idx = blockIdx.x * blockDim.x + threadIdx.x;
    if (idx >= npad * 512) return;
    int i = idx >> 9, c = idx & 511;
    int k = c >> 5, e = c & 31;
    float acc = 0.0f;
    if (i < n) {
        const float* Wk = W + (size_t)k * 1024;
        const float* ai = a + (size_t)i * 32;
        for (int d = 0; d < 32; ++d) acc += ai[d] * Wk[d * 32 + e];
    }
    out[idx] = f2bf(acc);
}

__global__ void a2bf_kernel(const float* __restrict__ a2, int n, int npad,
                            unsigned short* __restrict__ out) {
    int idx = blockIdx.x * blockDim.x + threadIdx.x;
    if (idx >= npad * 32) return;
    int i = idx >> 5, e = idx & 31;
    float v = (i < n) ? a2[(size_t)i * 32 + e] : 0.0f;
    out[idx] = f2bf(v);
}

// ---------------- fused GED matrices: WMMA bf16 16x16x32 ----------------
// out_map[i,j] = sum_k vmap[k]*relu(T1m_k[i,:] . a2[j,:])  (same for cost)
// one wave = one 16-row M tile x four 16-col N tiles; B fragments reused
// across all 16 channels x 2 matrices. Channel loop fully unrolled with
// pre-loaded channel weights (no in-loop scalar loads / kmcnt waits).
__global__ __launch_bounds__(128)
void ged_wmma_kernel(const unsigned short* __restrict__ T1c,
                     const unsigned short* __restrict__ T1m,
                     const unsigned short* __restrict__ A2,
                     const float* __restrict__ vcost,
                     const float* __restrict__ vmap,
                     float* __restrict__ map_out,
                     float* __restrict__ cost_out,
                     int n1, int n2) {
    const int wave = threadIdx.x >> 5;
    const int lane = threadIdx.x & 31;
    const int mtile = blockIdx.x * 4 + wave;   // 16 rows each
    const int ngrp  = blockIdx.y;              // 64 cols each
    const int lrow  = lane & 15;
    const int khalf = lane >> 4;

    // preload channel combination weights (uniform -> SGPRs, constant-indexed)
    float vc[16], vm[16];
#pragma unroll
    for (int ch = 0; ch < 16; ++ch) { vc[ch] = vcost[ch]; vm[ch] = vmap[ch]; }

    // B fragments: column = a2 row, 16 contiguous K per lane
    v16bf bfrag[4];
#pragma unroll
    for (int t = 0; t < 4; ++t) {
        int col = ngrp * 64 + t * 16 + lrow;
        bfrag[t] = *(const v16bf*)(A2 + (size_t)col * 32 + khalf * 16);
    }

    const v8f z8 = {0.f, 0.f, 0.f, 0.f, 0.f, 0.f, 0.f, 0.f};
    v8f accC[4], accM[4];
#pragma unroll
    for (int t = 0; t < 4; ++t) { accC[t] = z8; accM[t] = z8; }

    const int arow = mtile * 16 + lrow;
    const unsigned short* arC = T1c + (size_t)arow * 512 + khalf * 16;
    const unsigned short* arM = T1m + (size_t)arow * 512 + khalf * 16;

#pragma unroll
    for (int ch = 0; ch < 16; ++ch) {
        // cost channel
        {
            v8bf lo = *(const v8bf*)(arC + ch * 32);
            v8bf hi = *(const v8bf*)(arC + ch * 32 + 8);
            v16bf a = __builtin_shufflevector(lo, hi, 0,1,2,3,4,5,6,7,8,9,10,11,12,13,14,15);
#pragma unroll
            for (int t = 0; t < 4; ++t) {
                v8f d = __builtin_amdgcn_wmma_f32_16x16x32_bf16(
                    false, a, false, bfrag[t], (short)0, z8, false, false);
#pragma unroll
                for (int r = 0; r < 8; ++r) accC[t][r] += vc[ch] * relu1(d[r]);
            }
        }
        // map channel
        {
            v8bf lo = *(const v8bf*)(arM + ch * 32);
            v8bf hi = *(const v8bf*)(arM + ch * 32 + 8);
            v16bf a = __builtin_shufflevector(lo, hi, 0,1,2,3,4,5,6,7,8,9,10,11,12,13,14,15);
#pragma unroll
            for (int t = 0; t < 4; ++t) {
                v8f d = __builtin_amdgcn_wmma_f32_16x16x32_bf16(
                    false, a, false, bfrag[t], (short)0, z8, false, false);
#pragma unroll
                for (int r = 0; r < 8; ++r) accM[t][r] += vm[ch] * relu1(d[r]);
            }
        }
    }

    // D layout: lanes 0-15 -> M=r, N=lane; lanes 16-31 -> M=8+r, N=lane-16
    const int mb = mtile * 16 + khalf * 8;
#pragma unroll
    for (int t = 0; t < 4; ++t) {
        int col = ngrp * 64 + t * 16 + lrow;
        if (col < n2) {
#pragma unroll
            for (int r = 0; r < 8; ++r) {
                int row = mb + r;
                if (row < n1) {
                    size_t o = (size_t)row * n2 + col;
                    map_out[o]  = accM[t][r];
                    cost_out[o] = accC[t][r];
                }
            }
        }
    }
}

// ---------------- softmax(mapm,axis=1)*cost, global sum ----------------
__global__ void softmax_reduce_kernel(const float* __restrict__ mapm,
                                      const float* __restrict__ cost,
                                      int n2, float* __restrict__ soft_sum) {
    __shared__ float red[256];
    const int row = blockIdx.x;
    const int tid = threadIdx.x;
    const float* mrow = mapm + (size_t)row * n2;
    const float* crow = cost + (size_t)row * n2;
    float m = -3.4e38f;
    for (int j = tid; j < n2; j += 256) m = fmaxf(m, mrow[j]);
    red[tid] = m; __syncthreads();
    for (int s = 128; s > 0; s >>= 1) { if (tid < s) red[tid] = fmaxf(red[tid], red[tid + s]); __syncthreads(); }
    m = red[0]; __syncthreads();
    float se = 0.0f, sec = 0.0f;
    for (int j = tid; j < n2; j += 256) {
        float e = expf(mrow[j] - m);
        se += e; sec += e * crow[j];
    }
    red[tid] = se; __syncthreads();
    for (int s = 128; s > 0; s >>= 1) { if (tid < s) red[tid] += red[tid + s]; __syncthreads(); }
    se = red[0]; __syncthreads();
    red[tid] = sec; __syncthreads();
    for (int s = 128; s > 0; s >>= 1) { if (tid < s) red[tid] += red[tid + s]; __syncthreads(); }
    if (tid == 0) atomicAdd(soft_sum, red[0] / se);
}

// ---------------- mlp logits + gumbel perturbation ----------------
__global__ void logits_pert_kernel(const float* __restrict__ a1, const float* __restrict__ p2,
                                   const float* __restrict__ W1, const float* __restrict__ b1,
                                   const float* __restrict__ W2, const float* __restrict__ b2,
                                   int n, float* __restrict__ pert) {
    int i = blockIdx.x * blockDim.x + threadIdx.x;
    if (i >= n) return;
    const float* ai = a1 + (size_t)i * 32;
    float logit = b2[0];
    for (int o = 0; o < 32; ++o) {
        float acc = b1[o];
        for (int d = 0; d < 32; ++d) acc += ai[d] * W1[d * 32 + o];
        for (int d = 0; d < 32; ++d) acc += p2[d] * W1[(32 + d) * 32 + o];
        logit += fmaxf(acc, 0.0f) * W2[o];
    }
    float u = hash_uniform((unsigned int)i);
    pert[i] = logit - logf(-logf(u));
}

// top-k by rank counting (strict total order via index tie-break)
__global__ void rank_topk_kernel(const float* __restrict__ pert, int n1, int n2,
                                 float* __restrict__ out_idx) {
    int i = blockIdx.x * blockDim.x + threadIdx.x;
    if (i >= n1) return;
    float v = pert[i];
    int rank = 0;
    for (int j = 0; j < n1; ++j) {
        float w = pert[j];
        rank += (w > v) || (w == v && j < i);
    }
    if (rank < n2) out_idx[rank] = (float)i;
}

// ---------------- tensor network + fc head + score ----------------
__global__ void final_kernel(const float* __restrict__ p1, const float* __restrict__ p2,
                             const float* __restrict__ tnW, const float* __restrict__ tnWb,
                             const float* __restrict__ tnb,
                             const float* __restrict__ fc1W, const float* __restrict__ fc1b,
                             const float* __restrict__ fc2W, const float* __restrict__ fc2b,
                             const float* __restrict__ fc3W, const float* __restrict__ fc3b,
                             const float* __restrict__ scW, const float* __restrict__ scb,
                             const float* __restrict__ soft_sum, const float* __restrict__ avg_v,
                             float* __restrict__ out) {
    if (threadIdx.x != 0 || blockIdx.x != 0) return;
    float s[16];
    for (int t = 0; t < 16; ++t) {
        float acc = 0.0f;
        for (int e = 0; e < 32; ++e) {
            float sc = 0.0f;
            for (int d = 0; d < 32; ++d) sc += p1[d] * tnW[d * 512 + e * 16 + t];
            acc += sc * p2[e];
        }
        float blk = 0.0f;
        for (int j = 0; j < 32; ++j) blk += tnWb[t * 64 + j] * p1[j];
        for (int j = 0; j < 32; ++j) blk += tnWb[t * 64 + 32 + j] * p2[j];
        s[t] = fmaxf(acc + blk + tnb[t], 0.0f);
    }
    float h1[16];
    for (int o = 0; o < 16; ++o) {
        float a = fc1b[o];
        for (int t = 0; t < 16; ++t) a += s[t] * fc1W[t * 16 + o];
        h1[o] = fmaxf(a, 0.0f);
    }
    float h2[8];
    for (int o = 0; o < 8; ++o) {
        float a = fc2b[o];
        for (int t = 0; t < 16; ++t) a += h1[t] * fc2W[t * 8 + o];
        h2[o] = fmaxf(a, 0.0f);
    }
    float h3[4];
    for (int o = 0; o < 4; ++o) {
        float a = fc3b[o];
        for (int t = 0; t < 8; ++t) a += h2[t] * fc3W[t * 4 + o];
        h3[o] = fmaxf(a, 0.0f);
    }
    float bias = scb[0];
    for (int c = 0; c < 4; ++c) bias += h3[c] * scW[c];
    float score = 1.0f / (1.0f + expf(-(soft_sum[0] + bias)));
    out[0] = score;
    out[1] = -logf(score) * avg_v[0];
}

// ---------------- host launcher ----------------
extern "C" void kernel_launch(void* const* d_in, const int* in_sizes, int n_in,
                              void* d_out, int out_size, void* d_ws, size_t ws_size,
                              hipStream_t stream) {
    (void)n_in; (void)out_size; (void)ws_size;
    const int LABELS = 29;
    const int n1 = in_sizes[4] / LABELS;
    const int n2 = in_sizes[5] / LABELS;
    const int E1 = in_sizes[2] / 2;
    const int E2 = in_sizes[3] / 2;
    const int MPAD = ((n1 + 63) / 64) * 64;
    const int NPAD = ((n2 + 63) / 64) * 64;

    // input pointers (setup_inputs dict order, params flattened in insertion order)
    const int*   ei1 = (const int*)d_in[2];
    const int*   ei2 = (const int*)d_in[3];
    const float* f1  = (const float*)d_in[4];
    const float* f2  = (const float*)d_in[5];
    const float* avg_v = (const float*)d_in[6];
    auto P = [&](int i) { return (const float*)d_in[i]; };
    // gin l (1..3): base 7 + (l-1)*7 : W1,b1,W2,b2,gamma,beta,eps
    const float* att_W  = P(28);
    const float* cost_W = P(29); const float* cost_v = P(30);
    const float* map_W  = P(31); const float* map_v  = P(32);
    const float* tn_W = P(33); const float* tn_Wb = P(34); const float* tn_b = P(35);
    const float* fc1W = P(36); const float* fc1b = P(37);
    const float* fc2W = P(38); const float* fc2b = P(39);
    const float* fc3W = P(40); const float* fc3b = P(41);
    const float* scW  = P(42); const float* scb  = P(43);
    const float* mlpW1 = P(44); const float* mlpb1 = P(45);
    const float* mlpW2 = P(46); const float* mlpb2 = P(47);

    // workspace carve-up
    char* ws = (char*)d_ws;
    size_t off = 0;
    auto take = [&](size_t bytes) -> void* {
        void* p = ws + off;
        off += (bytes + 255) & ~(size_t)255;
        return p;
    };
    float* agg   = (float*)take((size_t)n1 * 128 * 4);
    float* hmid  = (float*)take((size_t)n1 * 128 * 4);
    float* hbuf  = (float*)take((size_t)n1 * 128 * 4);
    float* xA    = (float*)take((size_t)n1 * 128 * 4);
    float* xB    = (float*)take((size_t)n1 * 128 * 4);
    float* a1b   = (float*)take((size_t)n1 * 32 * 4);
    float* a2b   = (float*)take((size_t)n2 * 32 * 4);
    float* stats = (float*)take(256 * 4);
    float* misc  = (float*)take(256 * 4);
    float* pert  = (float*)take((size_t)n1 * 4);
    unsigned short* T1c  = (unsigned short*)take((size_t)MPAD * 512 * 2);
    unsigned short* T1m  = (unsigned short*)take((size_t)MPAD * 512 * 2);
    unsigned short* A2bf = (unsigned short*)take((size_t)NPAD * 32 * 2);
    float* cost  = (float*)take((size_t)n1 * (size_t)n2 * 4);

    float* soft_sum = misc + 0;
    float* p1   = misc + 32;
    float* p2   = misc + 64;
    float* cs1  = misc + 96;
    float* cs2  = misc + 128;
    float* ctx1 = misc + 160;
    float* ctx2 = misc + 192;

    float* out      = (float*)d_out;
    float* mapm     = out + 2;
    float* idx_out  = out + 2 + (size_t)n1 * (size_t)n2;

    const int TB = 256;
    auto blk = [&](long long n) { return (unsigned)((n + TB - 1) / TB); };

    // zero misc accumulators
    zero_f32<<<1, 256, 0, stream>>>(misc, 256);

    // ---- GIN conv pass for one graph ----
    auto conv_pass = [&](const float* feats, const int* ei, int E, int n, float* a_out) {
        const float* xcur = feats;
        int din = LABELS;
        const int douts[3] = {128, 64, 32};
        float* xnext[3] = {xA, xB, a_out};
        for (int l = 0; l < 3; ++l) {
            int base = 7 + l * 7;
            const float* W1 = P(base + 0); const float* b1 = P(base + 1);
            const float* W2 = P(base + 2); const float* b2 = P(base + 3);
            const float* ga = P(base + 4); const float* be = P(base + 5);
            const float* ep = P(base + 6);
            int dout = douts[l];
            zero_f32<<<blk((long long)n * din), TB, 0, stream>>>(agg, n * din);
            scatter_add_kernel<<<blk((long long)E * din), TB, 0, stream>>>(xcur, ei, agg, E, din);
            gin_lin1_kernel<<<blk((long long)n * dout), TB, 0, stream>>>(xcur, agg, ep, W1, b1, hmid, n, din, dout);
            zero_f32<<<1, 256, 0, stream>>>(stats, 2 * dout);
            gin_lin2_kernel<<<blk((long long)n * dout), TB, 0, stream>>>(hmid, W2, b2, hbuf, stats, n, dout, dout);
            gin_bn_kernel<<<blk((long long)n * dout), TB, 0, stream>>>(hbuf, stats, ga, be, xnext[l], n, dout, l < 2 ? 1 : 0);
            xcur = xnext[l];
            din = dout;
        }
    };
    conv_pass(f1, ei1, E1, n1, a1b);
    conv_pass(f2, ei2, E2, n2, a2b);

    // ---- attention pooling ----
    colsum_kernel<<<blk((long long)n1 * 32), TB, 0, stream>>>(a1b, n1, cs1);
    ctx_kernel<<<1, 32, 0, stream>>>(cs1, att_W, 1.0f / (float)n1, ctx1);
    pool_kernel<<<blk(n1), TB, 0, stream>>>(a1b, ctx1, n1, p1);
    colsum_kernel<<<blk((long long)n2 * 32), TB, 0, stream>>>(a2b, n2, cs2);
    ctx_kernel<<<1, 32, 0, stream>>>(cs2, att_W, 1.0f / (float)n2, ctx2);
    pool_kernel<<<blk(n2), TB, 0, stream>>>(a2b, ctx2, n2, p2);

    // ---- GED matrices via WMMA ----
    t1_kernel<<<blk((long long)MPAD * 512), TB, 0, stream>>>(a1b, cost_W, n1, MPAD, T1c);
    t1_kernel<<<blk((long long)MPAD * 512), TB, 0, stream>>>(a1b, map_W, n1, MPAD, T1m);
    a2bf_kernel<<<blk((long long)NPAD * 32), TB, 0, stream>>>(a2b, n2, NPAD, A2bf);

    dim3 gedGrid(MPAD / 64, NPAD / 64);
    ged_wmma_kernel<<<gedGrid, 128, 0, stream>>>(T1c, T1m, A2bf, cost_v, map_v, mapm, cost, n1, n2);

    // ---- softmax(map)*cost sum ----
    softmax_reduce_kernel<<<n1, 256, 0, stream>>>(mapm, cost, n2, soft_sum);

    // ---- gumbel top-k mask indices ----
    logits_pert_kernel<<<blk(n1), TB, 0, stream>>>(a1b, p2, mlpW1, mlpb1, mlpW2, mlpb2, n1, pert);
    rank_topk_kernel<<<blk(n1), TB, 0, stream>>>(pert, n1, n2, idx_out);

    // ---- tensor net + fc head + final score ----
    final_kernel<<<1, 32, 0, stream>>>(p1, p2, tn_W, tn_Wb, tn_b,
                                       fc1W, fc1b, fc2W, fc2b, fc3W, fc3b, scW, scb,
                                       soft_sum, avg_v, out);
}
